// GMesReduce_45208825757980
// MI455X (gfx1250) — matrix operand order repaired and checked by reading, not compiled, not run on hardware
//
#include <hip/hip_runtime.h>
#include <hip/hip_bf16.h>
#include <math.h>

#define NNODES 50000
#define DIM 128
#define NITERS 20
#define BIGDEG 5000.0f
#define TILES (NNODES / 16)   // 3125, exact

typedef __attribute__((ext_vector_type(16))) _Float16 v16h;
typedef __attribute__((ext_vector_type(8)))  _Float16 v8h;
typedef __attribute__((ext_vector_type(8)))  float    v8f;

// ---------------------------------------------------------------- utilities
__device__ __forceinline__ unsigned hash3(unsigned a, unsigned b, unsigned c) {
    unsigned h = a * 0x9E3779B9u ^ b * 0x85EBCA6Bu ^ c * 0xC2B2AE35u ^ 0x27220A95u;
    h ^= h >> 16; h *= 0x7FEB352Du;
    h ^= h >> 15; h *= 0x846CA68Bu;
    h ^= h >> 16;
    return h;
}

// ------------------------------------------------------------- init kernels
__global__ void init_kernel(const float* __restrict__ in, float* __restrict__ out,
                            float* __restrict__ deg, int nElem, int nNodes) {
    int i = blockIdx.x * blockDim.x + threadIdx.x;
    if (i < nElem) out[i] = in[i];
    if (i < nNodes) deg[i] = 0.0f;
}

__global__ void deg_kernel(const int* __restrict__ dst, float* __restrict__ deg, int nEdges) {
    int e = blockIdx.x * blockDim.x + threadIdx.x;
    if (e < nEdges) atomicAdd(&deg[dst[e]], 1.0f);
}

__global__ void zero_kernel(float4* __restrict__ msg, float* __restrict__ normsq, int nVec) {
    int i = blockIdx.x * blockDim.x + threadIdx.x;
    if (i < nVec) msg[i] = make_float4(0.f, 0.f, 0.f, 0.f);
    if (i == 0) *normsq = 0.0f;
}

// --------------------------------------------------------------- scatter
// msg[dst] += rep[src]; 32 threads/edge, float4 gather + 4 f32 atomics each.
// 25.6 MB node table is L2-resident on MI455X (192 MB L2): L2/atomic bound.
__global__ void scatter_kernel(const float* __restrict__ rep, const int* __restrict__ src,
                               const int* __restrict__ dst, float* __restrict__ msg,
                               int nEdges) {
    long long tid = (long long)blockIdx.x * blockDim.x + threadIdx.x;
    int e = (int)(tid >> 5);
    if (e >= nEdges) return;
    int q = ((int)tid & 31) * 4;
    const float4 v = *(const float4*)(rep + (size_t)src[e] * DIM + q);
    float* o = msg + (size_t)dst[e] * DIM + q;
    atomicAdd(o + 0, v.x);
    atomicAdd(o + 1, v.y);
    atomicAdd(o + 2, v.z);
    atomicAdd(o + 3, v.w);
}

// --------------------------------------------------------------- update
// One wave32 per 16-node tile; negatives shared per tile -> both contrastive
// einsums are exact 16x16 WMMA GEMMs:
//   GEMM1: ratio_neg(16x16) = repTile(16x128) * negT(128x16)   -> 4x v_wmma
//   GEMM2: negSum(16x128)   = P(16x32pad)    * Neg(32padx128)  -> 8x v_wmma
// All operand builds are branch-free (masks / address selects) so loads stay
// vectorized (global_load_b128 / ds_load_b128) with EXEC all-ones throughout.
__global__ void __launch_bounds__(32) update_kernel(
        const float* __restrict__ rep, const float* __restrict__ msg,
        const float* __restrict__ deg, float* __restrict__ grad,
        float* __restrict__ normsq, int iter) {
    __shared__ _Float16 negT[DIM + 16][16]; // transposed negs; rows DIM.. are zero pad
    __shared__ float ratioLDS[16][16];      // GEMM1 C tile spill
    __shared__ float pLDS[16][20];          // softmax probs, cols 11..19 zero
    __shared__ float part[32];              // ratio0 dot partials
    __shared__ float degL[16];              // per-node degree

    const int lane = threadIdx.x;
    const int col  = lane & 15;
    const int half = lane >> 4;
    const int m0   = blockIdx.x * 16;

    if (lane < 16) {
        degL[lane] = deg[m0 + lane];
        v8h z = {};
        *(v8h*)&negT[DIM + lane][0] = z;    // zero pad rows (GEMM2 K=16..31)
        *(v8h*)&negT[DIM + lane][8] = z;
    }
    __syncthreads();

    // Shared negative index for this tile's column `col` (branch-free mask)
    unsigned h = hash3((unsigned)iter, (unsigned)blockIdx.x, (unsigned)col);
    const int   nj  = (col < 10) ? (2 + (int)(h % (unsigned)(NNODES - 2))) : 0;
    const float msk = (col < 10) ? 1.0f : 0.0f;
    const float* negrow = rep + (size_t)nj * DIM;
    const float* arow   = rep + (size_t)(m0 + col) * DIM;

    // ---- GEMM1: ratio_neg over K=128 in 4 chunks of 32
    v8f acc = {0.f, 0.f, 0.f, 0.f, 0.f, 0.f, 0.f, 0.f};
    #pragma unroll
    for (int c = 0; c < 4; ++c) {
        const int k0 = c * 32;
        // A: lane holds row M=col; half 0 -> K k0+[0..15]; half 1 -> K k0+[16..31]
        float af[16];
        const float4* ap0 = (const float4*)(arow + k0 + half * 16);
        const float4* ap1 = (const float4*)(arow + k0 + half * 16 + 8);
        *(float4*)&af[0]  = ap0[0];
        *(float4*)&af[4]  = ap0[1];
        *(float4*)&af[8]  = ap1[0];
        *(float4*)&af[12] = ap1[1];
        v16h a;
        #pragma unroll
        for (int i = 0; i < 16; ++i) a[i] = (_Float16)af[i];

        // B: lane holds column N=col, K rows kb..kb+15 (unconditional b128 loads)
        const int kb = k0 + half * 16;
        float bf[16];
        const float4* np = (const float4*)(negrow + kb);
        *(float4*)&bf[0]  = np[0];
        *(float4*)&bf[4]  = np[1];
        *(float4*)&bf[8]  = np[2];
        *(float4*)&bf[12] = np[3];
        v16h b;
        #pragma unroll
        for (int i = 0; i < 16; ++i) {
            _Float16 hv = (_Float16)(bf[i] * msk);
            b[i] = hv;
            negT[kb + i][col] = hv;         // transpose-stage for GEMM2
        }
        acc = __builtin_amdgcn_wmma_f32_16x16x32_f16(false, a, false, b,
                                                     (short)0, acc, false, false);
    }

    // ---- ratio0 = <mean_msg, rep> per node; each lane covers 64 dims
    {
        const int m = m0 + col;
        const float invd = 1.0f / fmaxf(degL[col], 1.0f);
        const float4* rp = (const float4*)(rep + (size_t)m * DIM + half * 64);
        const float4* mp = (const float4*)(msg + (size_t)m * DIM + half * 64);
        float s = 0.0f;
        #pragma unroll
        for (int i = 0; i < 16; ++i) {
            float4 r4 = rp[i], m4 = mp[i];
            s += r4.x * m4.x + r4.y * m4.y + r4.z * m4.z + r4.w * m4.w;
        }
        part[lane] = s * invd;
    }
    __syncthreads();

    // ---- spill GEMM1 C tile: VGPR r holds row M = r + 8*half, column N = col
    #pragma unroll
    for (int r = 0; r < 8; ++r) ratioLDS[r + 8 * half][col] = acc[r];
    __syncthreads();

    // ---- 11-way softmax, one node per lane (lanes 0..15)
    if (lane < 16) {
        float v[11];
        v[0] = part[lane] + part[lane + 16];
        #pragma unroll
        for (int j = 0; j < 10; ++j) v[1 + j] = ratioLDS[lane][j];
        float mx = v[0];
        #pragma unroll
        for (int k = 1; k < 11; ++k) mx = fmaxf(mx, v[k]);
        float p[11], sum = 0.0f;
        #pragma unroll
        for (int k = 0; k < 11; ++k) { p[k] = __expf(v[k] - mx); sum += p[k]; }
        float inv = 1.0f / sum;
        #pragma unroll
        for (int k = 0; k < 11; ++k) pLDS[lane][k] = p[k] * inv;
        #pragma unroll
        for (int k = 11; k < 20; ++k) pLDS[lane][k] = 0.0f;
    }
    __syncthreads();

    // ---- GEMM2 A: A2[m][K] = p_{K+1}(node m) for K<10; branch-free indexed read
    v16h a2;
    #pragma unroll
    for (int i = 0; i < 8; ++i) {
        a2[i]     = (_Float16)(half ? 0.0f : pLDS[col][i + 1]);      // K = i (half 1: pad)
        a2[8 + i] = (_Float16)(half ? 0.0f : pLDS[col][8 + i + 1]);  // K = 8+i (half 1: pad)
    }

    // Per-node scalars cached in registers for the grad pass
    float invdL[8], p0L[8], eligL[8];
    #pragma unroll
    for (int r = 0; r < 8; ++r) {
        const int ml = r + 8 * half;
        float dg = degL[ml];
        invdL[r] = 1.0f / fmaxf(dg, 1.0f);
        eligL[r] = (dg >= 2.0f && dg <= BIGDEG) ? 1.0f : 0.0f;
        p0L[r]   = pLDS[ml][0];
    }

    // ---- GEMM2 + grad: 8 output chunks of 16 dims
    float nsum = 0.0f;
    for (int n0 = 0; n0 < DIM; n0 += 16) {
        // B2 row for this lane: half 0 -> negT[n0+col], half 1 -> zero pad row
        const int row = half ? (DIM + col) : (n0 + col);
        v8h b_lo = *(const v8h*)&negT[row][0];   // one contiguous 32B row: 2x b128
        v8h b_hi = *(const v8h*)&negT[row][8];
        v16h b2 = __builtin_shufflevector(b_lo, b_hi,
                                          0, 1, 2, 3, 4, 5, 6, 7,
                                          8, 9, 10, 11, 12, 13, 14, 15);
        v8f acc2 = {0.f, 0.f, 0.f, 0.f, 0.f, 0.f, 0.f, 0.f};
        acc2 = __builtin_amdgcn_wmma_f32_16x16x32_f16(false, a2, false, b2,
                                                      (short)0, acc2, false, false);
        const int d = n0 + col;
        #pragma unroll
        for (int r = 0; r < 8; ++r) {
            const int m = m0 + r + 8 * half;
            float mean = msg[(size_t)m * DIM + d] * invdL[r];
            float g = eligL[r] * (acc2[r] + (p0L[r] - 1.0f) * mean);
            grad[(size_t)m * DIM + d] = g;
            nsum += g * g;
        }
    }

    // ---- wave-reduce grad^2 and accumulate global norm
    #pragma unroll
    for (int off = 16; off > 0; off >>= 1) nsum += __shfl_down(nsum, off, 32);
    if (lane == 0) atomicAdd(normsq, nsum);
}

// --------------------------------------------------------------- finalize
// new_rep = big ? mean_msg : rep - coef*grad  (grad already masked by eligible)
__global__ void finalize_kernel(float* __restrict__ rep, const float* __restrict__ msg,
                                const float* __restrict__ grad, const float* __restrict__ deg,
                                const float* __restrict__ normsq, int nNodes) {
    int tid = blockIdx.x * blockDim.x + threadIdx.x;   // one thread per (node, 4 dims)
    int m = tid >> 5;
    if (m >= nNodes) return;
    int q = (tid & 31) * 4;
    float d  = deg[m];
    float ds = fmaxf(d, 1.0f);
    bool  big = d > BIGDEG;
    float coef = fminf(1.0f, 2.0f / (sqrtf(*normsq) + 1e-6f));
    size_t off = (size_t)m * DIM + q;
    float4 r  = *(const float4*)(rep + off);
    float4 ms = *(const float4*)(msg + off);
    float4 g  = *(const float4*)(grad + off);
    float4 o;
    if (big) {
        o = make_float4(ms.x / ds, ms.y / ds, ms.z / ds, ms.w / ds);
    } else {
        o = make_float4(r.x - coef * g.x, r.y - coef * g.y,
                        r.z - coef * g.z, r.w - coef * g.w);
    }
    *(float4*)(rep + off) = o;
}

// --------------------------------------------------------------- launcher
extern "C" void kernel_launch(void* const* d_in, const int* in_sizes, int n_in,
                              void* d_out, int out_size, void* d_ws, size_t ws_size,
                              hipStream_t stream) {
    const float* repre = (const float*)d_in[0];
    const int*   src   = (const int*)d_in[1];
    const int*   dst   = (const int*)d_in[2];
    const int nEdges   = in_sizes[1];

    float* rep    = (float*)d_out;                 // working representation (in-place)
    float* msg    = (float*)d_ws;                  // N*D
    float* grad   = msg  + (size_t)NNODES * DIM;   // N*D
    float* deg    = grad + (size_t)NNODES * DIM;   // N
    float* normsq = deg  + NNODES;                 // 1

    const int nElem = NNODES * DIM;
    const int nVec  = nElem / 4;

    init_kernel<<<(nElem + 255) / 256, 256, 0, stream>>>(repre, rep, deg, nElem, NNODES);
    deg_kernel<<<(nEdges + 255) / 256, 256, 0, stream>>>(dst, deg, nEdges);

    const long long scatterThreads = (long long)nEdges * 32;
    const int scatterBlocks = (int)((scatterThreads + 255) / 256);
    const int finThreads = NNODES * 32;

    for (int it = 0; it < NITERS; ++it) {
        zero_kernel<<<(nVec + 255) / 256, 256, 0, stream>>>((float4*)msg, normsq, nVec);
        scatter_kernel<<<scatterBlocks, 256, 0, stream>>>(rep, src, dst, msg, nEdges);
        update_kernel<<<TILES, 32, 0, stream>>>(rep, msg, deg, grad, normsq, it);
        finalize_kernel<<<(finThreads + 255) / 256, 256, 0, stream>>>(rep, msg, grad, deg,
                                                                      normsq, NNODES);
    }
}
